// MAB_48412871360901
// MI455X (gfx1250) — compile-verified, matrix operand order checked by reading, not run on hardware
//
#include <hip/hip_runtime.h>

typedef __attribute__((ext_vector_type(16))) _Float16 v16h;
typedef __attribute__((ext_vector_type(8)))  _Float16 v8h;
typedef __attribute__((ext_vector_type(2)))  _Float16 v2h;
typedef __attribute__((ext_vector_type(8)))  float    v8f;
typedef int v4i_b __attribute__((vector_size(16)));   // builtin's pointee type

#define BB   16
#define NQ   1024
#define NK   1024
#define DIM  512
#define HH   8
#define HD   64
#define MROWS (BB*NQ)                  // 16384
#define SCALE 0.044194173824159216f   // 1/sqrt(512)

// ---------------------------------------------------------------------------
// CDNA5 async global->LDS copy (ASYNCcnt-tracked), with synchronous fallback.
// ---------------------------------------------------------------------------
#if defined(__HIP_DEVICE_COMPILE__) && \
    __has_builtin(__builtin_amdgcn_global_load_async_to_lds_b128)
#define USE_ASYNC_LDS 1
#else
#define USE_ASYNC_LDS 0
#endif

__device__ __forceinline__ void cp16_g2l(_Float16* lds, const _Float16* g) {
#if USE_ASYNC_LDS
  __builtin_amdgcn_global_load_async_to_lds_b128(
      (__attribute__((address_space(1))) v4i_b*)g,
      (__attribute__((address_space(3))) v4i_b*)lds, 0, 0);
#else
  *(v8h*)lds = *(const v8h*)g;
#endif
}

__device__ __forceinline__ void cp_wait() {
#if USE_ASYNC_LDS
#if __has_builtin(__builtin_amdgcn_s_wait_asynccnt)
  __builtin_amdgcn_s_wait_asynccnt(0);
#else
  asm volatile("s_wait_asynccnt 0" ::: "memory");
#endif
#endif
}

// ---------------------------------------------------------------------------
// WMMA helpers
// ---------------------------------------------------------------------------
__device__ __forceinline__ v8f wmma_f16(v16h a, v16h b, v8f c) {
  return __builtin_amdgcn_wmma_f32_16x16x32_f16(
      false, a, false, b, (short)0, c, false, false);
}

__device__ __forceinline__ v8f zero8() {
  v8f z = {0.f,0.f,0.f,0.f,0.f,0.f,0.f,0.f};
  return z;
}

// 16x32 f16 fragment from LDS (row-major, `stride` halves between rows).
// Lane l: row = l&15, kb = (l>=16)*8; elems 0..7 = K kb..kb+7, 8..15 = kb+16..+23.
__device__ __forceinline__ v16h ld_frag(const _Float16* base, int stride) {
  const int lane = threadIdx.x & 31;
  const _Float16* p = base + (lane & 15) * stride + ((lane >> 4) << 3);
  v8h lo = *(const v8h*)(p);
  v8h hi = *(const v8h*)(p + 16);
  v16h f;
#pragma unroll
  for (int i = 0; i < 8; ++i) { f[i] = lo[i]; f[i + 8] = hi[i]; }
  return f;
}

// ---------------------------------------------------------------------------
// Prepass A: elementwise f32 -> f16 (8 elems/thread, vectorized)
// ---------------------------------------------------------------------------
__global__ __launch_bounds__(256)
void cvt_f16_mab(const float* __restrict__ src, _Float16* __restrict__ dst) {
  const int i = blockIdx.x * 256 + threadIdx.x;
  const float4* s = (const float4*)src;
  float4 a = s[2 * i], b = s[2 * i + 1];
  v8h o;
  o[0]=(_Float16)a.x; o[1]=(_Float16)a.y; o[2]=(_Float16)a.z; o[3]=(_Float16)a.w;
  o[4]=(_Float16)b.x; o[5]=(_Float16)b.y; o[6]=(_Float16)b.z; o[7]=(_Float16)b.w;
  *(v8h*)(dst + (size_t)i * 8) = o;
}

// Prepass B: W[k][n] f32 -> Wt[n][k] f16 (transposed, n-major for B-frags)
__global__ __launch_bounds__(256)
void cvt_wt_mab(const float* __restrict__ W, _Float16* __restrict__ Wt) {
  const int k = blockIdx.x;
  const int n = blockIdx.y * 256 + threadIdx.x;
  Wt[n * DIM + k] = (_Float16)W[k * DIM + n];
}

// ---------------------------------------------------------------------------
// GEMM: C[M,512] = A16[M,512] @ Wt16[n][k] + bias, 64x128 blocks.
// Whole A panel (64x512) + B panel (128x512) async-staged into LDS once,
// then 16 barrier-free WMMA k-steps.
// MODE 0: out16 = f16(C), optional out32 = C     MODE 1: outZ = resid + relu(C)
// ---------------------------------------------------------------------------
#define GSTR 520   // LDS row stride in halves (1040 B, 16B-aligned, padded)

template <int MODE>
__global__ __launch_bounds__(256)
void gemm512_mab(const _Float16* __restrict__ A16, const _Float16* __restrict__ Wt,
                 const float* __restrict__ bias,
                 _Float16* __restrict__ out16, float* __restrict__ out32,
                 const float* __restrict__ resid, float* __restrict__ outZ) {
  extern __shared__ char smem[];
  _Float16* sA = (_Float16*)smem;       // 64  x GSTR
  _Float16* sB = sA + 64 * GSTR;        // 128 x GSTR

  const int tid  = threadIdx.x;
  const int row0 = blockIdx.x * 64;
  const int col0 = blockIdx.y * 128;
  const int wave = tid >> 5;
  const int lane = tid & 31;
  const int mw   = wave & 3;
  const int nh   = wave >> 2;

  // stage full panels (16B chunks, ASYNCcnt-tracked)
  const _Float16* Ap = A16 + (size_t)row0 * DIM;
#pragma unroll
  for (int e = tid; e < 64 * 64; e += 256) {
    const int r = e >> 6, c = (e & 63) << 3;
    cp16_g2l(sA + r * GSTR + c, Ap + r * DIM + c);
  }
#pragma unroll
  for (int e = tid; e < 128 * 64; e += 256) {
    const int r = e >> 6, c = (e & 63) << 3;
    cp16_g2l(sB + r * GSTR + c, Wt + (col0 + r) * DIM + c);
  }
  cp_wait();
  __syncthreads();

  v8f acc[4] = {zero8(), zero8(), zero8(), zero8()};
#pragma unroll 4
  for (int kt = 0; kt < 16; ++kt) {
    const int k0 = kt * 32;
    v16h a = ld_frag(sA + mw * 16 * GSTR + k0, GSTR);
#pragma unroll
    for (int j = 0; j < 4; ++j) {
      v16h b = ld_frag(sB + (nh * 4 + j) * 16 * GSTR + k0, GSTR);
      acc[j] = wmma_f16(a, b, acc[j]);
    }
  }

  const int rhalf = (lane >> 4) << 3;
#pragma unroll
  for (int j = 0; j < 4; ++j) {
#pragma unroll
    for (int r = 0; r < 8; ++r) {
      const int grow = row0 + mw * 16 + rhalf + r;
      const int gcol = col0 + (nh * 4 + j) * 16 + (lane & 15);
      const float val = acc[j][r] + bias[gcol];
      const int idx = grow * DIM + gcol;
      if (MODE == 0) {
        out16[idx] = (_Float16)val;
        if (out32) out32[idx] = val;
      } else {
        outZ[idx] = resid[idx] + fmaxf(val, 0.f);
      }
    }
  }
}

// ---------------------------------------------------------------------------
// Attention for one (batch, head, 64-query tile); full 64x1024 score strip
// in LDS (f16). X += softmax(q k^T / sqrt(512)) @ v   (X holds q: residual).
// ---------------------------------------------------------------------------
__global__ __launch_bounds__(256)
void attn64_mab(const _Float16* __restrict__ q16, const _Float16* __restrict__ k16,
                const _Float16* __restrict__ v16, float* __restrict__ X) {
  extern __shared__ char smem[];
  _Float16* sP   = (_Float16*)smem;            // 64 x 1024 scores/probs
  _Float16* sQ   = sP + 64 * 1024;             // 64 x 64, stride 72
  _Float16* sKV  = sQ + 64 * 72;               // 64 x 64, stride 72
  float*    rsum = (float*)(sKV + 64 * 72);    // 64

  const int tid  = threadIdx.x;
  const int wave = tid >> 5;
  const int lane = tid & 31;
  const int qt   = blockIdx.x & 15;
  const int h    = (blockIdx.x >> 4) & 7;
  const int b    = blockIdx.x >> 7;

  const _Float16* gq = q16 + ((b * NQ + qt * 64) * DIM + h * HD);
  const _Float16* gk = k16 + (b * NK * DIM + h * HD);
  const _Float16* gv = v16 + (b * NK * DIM + h * HD);

  // stage q tile (async)
#pragma unroll
  for (int e = tid; e < 64 * 8; e += 256) {
    const int r = e >> 3, c = (e & 7) << 3;
    cp16_g2l(sQ + r * 72 + c, gq + r * DIM + c);
  }

  const int mw = wave & 3;
  const int nh = wave >> 2;
  const int rhalf = (lane >> 4) << 3;

  // ---- phase 1: S strip ---------------------------------------------------
  for (int kt = 0; kt < 16; ++kt) {
    __syncthreads();
#pragma unroll
    for (int e = tid; e < 64 * 8; e += 256) {       // K tile [key][dim], async
      const int r = e >> 3, c = (e & 7) << 3;
      cp16_g2l(sKV + r * 72 + c, gk + (kt * 64 + r) * DIM + c);
    }
    cp_wait();
    __syncthreads();
#pragma unroll
    for (int j = 0; j < 2; ++j) {
      const int nt = nh * 2 + j;
      v8f acc = zero8();
#pragma unroll
      for (int ks = 0; ks < 64; ks += 32) {
        v16h a  = ld_frag(sQ + mw * 16 * 72 + ks, 72);
        v16h bb = ld_frag(sKV + nt * 16 * 72 + ks, 72);   // n=key, k=dim
        acc = wmma_f16(a, bb, acc);
      }
#pragma unroll
      for (int r = 0; r < 8; ++r) {
        const int qrow = mw * 16 + rhalf + r;
        const int kcol = kt * 64 + nt * 16 + (lane & 15);
        sP[qrow * 1024 + kcol] = (_Float16)(acc[r] * SCALE);
      }
    }
  }
  __syncthreads();

  // ---- phase 2: row softmax (unnormalized exp stays in sP) ----------------
  for (int i = 0; i < 8; ++i) {
    const int row = wave * 8 + i;
    _Float16* pr = sP + row * 1024;
    float m = -1e30f;
#pragma unroll
    for (int j = 0; j < 32; ++j) m = fmaxf(m, (float)pr[lane + j * 32]);
#pragma unroll
    for (int o = 16; o > 0; o >>= 1) m = fmaxf(m, __shfl_xor(m, o, 32));
    float s = 0.f;
#pragma unroll
    for (int j = 0; j < 32; ++j) {
      const int c = lane + j * 32;
      float e = __expf((float)pr[c] - m);
      pr[c] = (_Float16)e;
      s += e;
    }
#pragma unroll
    for (int o = 16; o > 0; o >>= 1) s += __shfl_xor(s, o, 32);
    if (lane == 0) rsum[row] = s;
  }
  __syncthreads();

  // ---- phase 3: O = P @ V -------------------------------------------------
  v8f accO[2] = {zero8(), zero8()};
  for (int kt = 0; kt < 16; ++kt) {
    __syncthreads();
    // V tile transposed [dim][key], key pairs packed into b32 LDS stores
    {
      const int kp  = tid >> 3;            // key pair 0..31
      const int seg = (tid & 7) << 3;      // dim segment
      v8h va = *(const v8h*)(gv + (kt * 64 + 2 * kp)     * DIM + seg);
      v8h vb = *(const v8h*)(gv + (kt * 64 + 2 * kp + 1) * DIM + seg);
#pragma unroll
      for (int i = 0; i < 8; ++i) {
        v2h p; p[0] = va[i]; p[1] = vb[i];
        *(v2h*)(sKV + (seg + i) * 72 + 2 * kp) = p;
      }
    }
    __syncthreads();
#pragma unroll
    for (int j = 0; j < 2; ++j) {
      const int nt = nh * 2 + j;
#pragma unroll
      for (int ks = 0; ks < 64; ks += 32) {
        v16h a  = ld_frag(sP + mw * 16 * 1024 + kt * 64 + ks, 1024); // k=key
        v16h bb = ld_frag(sKV + nt * 16 * 72 + ks, 72);              // n=dim
        accO[j] = wmma_f16(a, bb, accO[j]);
      }
    }
  }

#pragma unroll
  for (int j = 0; j < 2; ++j) {
#pragma unroll
    for (int r = 0; r < 8; ++r) {
      const int qrow = mw * 16 + rhalf + r;
      const int dcol = (nh * 2 + j) * 16 + (lane & 15);
      float* px = X + ((b * NQ + qt * 64 + qrow) * DIM + h * HD + dcol);
      *px += accO[j][r] / rsum[qrow];
    }
  }
}

// ---------------------------------------------------------------------------
// Rowwise layernorm over 512; one wave per row, contiguous 16 elems per lane.
// ---------------------------------------------------------------------------
__global__ __launch_bounds__(256)
void ln512_mab(const float* __restrict__ Xin, const float* __restrict__ g,
               const float* __restrict__ bta, float* __restrict__ out32,
               _Float16* __restrict__ out16) {
  const int wave = threadIdx.x >> 5;
  const int lane = threadIdx.x & 31;
  const int row  = blockIdx.x * 8 + wave;
  const int c0   = lane * 16;
  const float* xr = Xin + row * DIM + c0;

  float x[16], s = 0.f;
#pragma unroll
  for (int q = 0; q < 4; ++q) {
    float4 t = *(const float4*)(xr + q * 4);
    x[q*4+0] = t.x; x[q*4+1] = t.y; x[q*4+2] = t.z; x[q*4+3] = t.w;
    s += t.x + t.y + t.z + t.w;
  }
#pragma unroll
  for (int o = 16; o > 0; o >>= 1) s += __shfl_xor(s, o, 32);
  const float mu = s * (1.f / 512.f);
  float v = 0.f;
#pragma unroll
  for (int j = 0; j < 16; ++j) { float d = x[j] - mu; v += d * d; }
#pragma unroll
  for (int o = 16; o > 0; o >>= 1) v += __shfl_xor(v, o, 32);
  const float rs = rsqrtf(v * (1.f / 512.f) + 1e-5f);

  float y[16];
#pragma unroll
  for (int j = 0; j < 16; ++j)
    y[j] = (x[j] - mu) * rs * g[c0 + j] + bta[c0 + j];
#pragma unroll
  for (int q = 0; q < 4; ++q) {
    float4 t; t.x=y[q*4+0]; t.y=y[q*4+1]; t.z=y[q*4+2]; t.w=y[q*4+3];
    *(float4*)(out32 + row * DIM + c0 + q * 4) = t;
  }
  if (out16) {
#pragma unroll
    for (int q = 0; q < 2; ++q) {
      v8h o;
#pragma unroll
      for (int i = 0; i < 8; ++i) o[i] = (_Float16)y[q * 8 + i];
      *(v8h*)(out16 + row * DIM + c0 + q * 8) = o;
    }
  }
}

// ---------------------------------------------------------------------------
extern "C" void kernel_launch(void* const* d_in, const int* in_sizes, int n_in,
                              void* d_out, int out_size, void* d_ws, size_t ws_size,
                              hipStream_t stream) {
  const float* Q  = (const float*)d_in[0];
  const float* K  = (const float*)d_in[1];
  const float* Wq = (const float*)d_in[2];
  const float* bq = (const float*)d_in[3];
  const float* Wk = (const float*)d_in[4];
  const float* bk = (const float*)d_in[5];
  const float* Wv = (const float*)d_in[6];
  const float* bv = (const float*)d_in[7];
  const float* Wo = (const float*)d_in[8];
  const float* bo = (const float*)d_in[9];
  const float* g0 = (const float*)d_in[10];
  const float* b0 = (const float*)d_in[11];
  const float* g1 = (const float*)d_in[12];
  const float* b1 = (const float*)d_in[13];
  float* out = (float*)d_out;

  // workspace carve (halves): q16|k16|v16|Qh|Kh|Wqt|Wkt|Wvt|Wot then X (f32)
  const size_t SZ = (size_t)MROWS * DIM;   // 8 Mi
  const size_t WS = (size_t)DIM * DIM;     // 256 Ki
  _Float16* q16 = (_Float16*)d_ws;
  _Float16* k16 = q16 + SZ;
  _Float16* v16 = k16 + SZ;
  _Float16* Qh  = v16 + SZ;
  _Float16* Kh  = Qh  + SZ;
  _Float16* Wqt = Kh  + SZ;
  _Float16* Wkt = Wqt + WS;
  _Float16* Wvt = Wkt + WS;
  _Float16* Wot = Wvt + WS;
  float*    X   = (float*)(Wot + WS);

  // prepass: convert activations + transpose-convert weights
  cvt_f16_mab<<<SZ / (8 * 256), 256, 0, stream>>>(Q, Qh);
  cvt_f16_mab<<<SZ / (8 * 256), 256, 0, stream>>>(K, Kh);
  dim3 wtg(DIM, DIM / 256);
  cvt_wt_mab<<<wtg, 256, 0, stream>>>(Wq, Wqt);
  cvt_wt_mab<<<wtg, 256, 0, stream>>>(Wk, Wkt);
  cvt_wt_mab<<<wtg, 256, 0, stream>>>(Wv, Wvt);
  cvt_wt_mab<<<wtg, 256, 0, stream>>>(Wo, Wot);

  dim3 ggrid(MROWS / 64, DIM / 128);
  const size_t gemm_lds = (size_t)(64 + 128) * GSTR * sizeof(_Float16); // ~200 KB
  // projections (q also kept as f32 in X for the residual)
  gemm512_mab<0><<<ggrid, 256, gemm_lds, stream>>>(Qh, Wqt, bq, q16, X,
                                                   nullptr, nullptr);
  gemm512_mab<0><<<ggrid, 256, gemm_lds, stream>>>(Kh, Wkt, bk, k16, nullptr,
                                                   nullptr, nullptr);
  gemm512_mab<0><<<ggrid, 256, gemm_lds, stream>>>(Kh, Wvt, bv, v16, nullptr,
                                                   nullptr, nullptr);
  // attention: X += attn
  const size_t attn_lds = (size_t)(64 * 1024 + 2 * 64 * 72) * sizeof(_Float16)
                        + 64 * sizeof(float);
  attn64_mab<<<BB * HH * (NQ / 64), 256, attn_lds, stream>>>(q16, k16, v16, X);
  // X = LN0(X) in place; f16 copy into q16 (q no longer needed)
  ln512_mab<<<MROWS / 8, 256, 0, stream>>>(X, g0, b0, X, q16);
  // X = X + relu(X @ Wo + bo) in place
  gemm512_mab<1><<<ggrid, 256, gemm_lds, stream>>>(q16, Wot, bo, nullptr, nullptr,
                                                   X, X);
  // out = LN1(X)
  ln512_mab<<<MROWS / 8, 256, 0, stream>>>(X, g1, b1, out, nullptr);
}
